// TransformerEncoderLayer_3874060501280
// MI455X (gfx1250) — compile-verified, matrix operand order checked by reading, not run on hardware
//
#include <hip/hip_runtime.h>
#include <hip/hip_bf16.h>

typedef __bf16 bf16_t;
typedef __attribute__((ext_vector_type(16))) __bf16 v16bf;
typedef __attribute__((ext_vector_type(8)))  __bf16 v8bf;
typedef __attribute__((ext_vector_type(8)))  float  v8f;
typedef __attribute__((ext_vector_type(4)))  unsigned int v4u;
typedef __attribute__((ext_vector_type(8)))  int v8i;
typedef __attribute__((ext_vector_type(4)))  int v4i;

#define LQ 1024
#define BQ 8
#define DM 768
#define NH 8
#define DH 96
#define DFF 3072
#define NREL 199      // 2*MAX_REL-1
#define NCOLS 224     // padded band width (14 WMMA n-tiles)
#define NT 14
#define TILE_ELEMS (128 * 40)

#if __has_builtin(__builtin_amdgcn_tensor_load_to_lds) && __has_builtin(__builtin_amdgcn_s_wait_tensorcnt)
#define USE_TDM 1
#else
#define USE_TDM 0
#endif

__device__ __forceinline__ v16bf cat8(v8bf lo, v8bf hi) {
  return __builtin_shufflevector(lo, hi, 0,1,2,3,4,5,6,7,8,9,10,11,12,13,14,15);
}

// Fragment loader for 16-bit WMMA A (or B built from a K-contiguous source,
// i.e. K^T / transposed-in-LDS B). Per ISA layout: lane&15 = M/N index,
// lane>>4 selects K-half; elem 0..7 -> K = kh*8+j, elem 8..15 -> K = 16+kh*8+j.
__device__ __forceinline__ v16bf load_frag_k_contig(const bf16_t* base, int stride, int lane) {
  int mn = lane & 15;
  int kh = (lane >> 4) << 3;
  const bf16_t* p = base + (size_t)mn * stride + kh;
  v8bf lo = *(const v8bf*)(p);
  v8bf hi = *(const v8bf*)(p + 16);
  return cat8(lo, hi);
}

__device__ __forceinline__ v8f wmma_bf16(v16bf a, v16bf b, v8f c) {
  return __builtin_amdgcn_wmma_f32_16x16x32_bf16(false, a, false, b, (short)0, c, false, false);
}

#if USE_TDM
// Issue a TDM load of a 2D bf16 tile (tileH rows x tileW cols) from a row-major
// matrix with row stride srcStride (elements) into LDS at lds_off, with LDS
// row padding controlled by pad interval/amount codes.
// D# layout per CDNA5 ISA 8.3/8.4. 6-arg builtin form:
//   (uint32x4 g0, int32x8 g1, int32x4 g2, int32x4 g3, int32x8 g4, i32 cpol)
__device__ __forceinline__ void tdm_load_tile_bf16(
    const bf16_t* src, unsigned int lds_off, int tileW, int tileH,
    int srcStride, int padIntervalCode, int padAmountCode)
{
  unsigned long long ga = (unsigned long long)(uintptr_t)src;
  v4u g0;
  g0[0] = 1u;                                        // count=1, user descriptor
  g0[1] = lds_off;                                   // LDS byte address
  g0[2] = (unsigned int)ga;                          // global_addr[31:0]
  g0[3] = (unsigned int)((ga >> 32) & 0x01FFFFFFu) | (2u << 30); // addr[56:32] | type=2
  int pad_en = (padAmountCode >= 0) ? 1 : 0;
  v8i g1;
  g1[0] = (1 << 16)                                  // data_size = 2 bytes
        | (pad_en << 20)
        | ((pad_en ? padIntervalCode : 0) << 22)
        | ((pad_en ? padAmountCode : 0) << 25);
  g1[1] = (tileW & 0xFFFF) << 16;                    // tensor_dim0[15:0]
  g1[2] = ((tileW >> 16) & 0xFFFF) | ((tileH & 0xFFFF) << 16);   // dim0 hi | dim1 lo
  g1[3] = ((tileH >> 16) & 0xFFFF) | ((tileW & 0xFFFF) << 16);   // dim1 hi | tile_dim0
  g1[4] = (tileH & 0xFFFF);                          // tile_dim1 | tile_dim2=0
  g1[5] = srcStride;                                 // tensor_dim0_stride[31:0]
  g1[6] = 0;
  g1[7] = 0;
  v4i gz4 = {0, 0, 0, 0};
  v8i gz8 = {0, 0, 0, 0, 0, 0, 0, 0};
  __builtin_amdgcn_tensor_load_to_lds(g0, g1, gz4, gz4, gz8, 0);
}
#endif

// ---------------- pack / convert kernels ----------------

__global__ void conv_f32_bf16(const float* __restrict__ in, bf16_t* __restrict__ out, int n) {
  int i = blockIdx.x * 256 + threadIdx.x;
  if (i < n) out[i] = (bf16_t)in[i];
}

// w_q/w_k/w_v: (H, DM, DH) -> combined (DM, 3*NH*DH = 2304) row-major
__global__ void pack_qkv_w(const float* __restrict__ wq, const float* __restrict__ wk,
                           const float* __restrict__ wv, bf16_t* __restrict__ out) {
  int i = blockIdx.x * 256 + threadIdx.x;
  if (i >= DM * 3 * DM) return;
  int col = i % (3 * DM), f = i / (3 * DM);
  int which = col / DM, rem = col % DM;
  int h = rem / DH, a = rem % DH;
  const float* w = (which == 0) ? wq : ((which == 1) ? wk : wv);
  out[i] = (bf16_t)w[((size_t)h * DM + f) * DH + a];
}

// ---------------- generic bf16 WMMA GEMM (double-buffered pipeline) ------
// C(MxN,f32 or bf16) = A(MxK bf16 rm) * B(KxN bf16 rm) [+bias][relu]
// block tile 128x128, 8 waves (4x2), wave tile 32x64, K-step 32.
// Stage i+1's A tile is DMA'd by the Tensor Data Mover and B tile's global
// loads are issued into registers BEFORE computing stage i, so the DMA and
// VMEM traffic overlap the WMMA chain; TENSORcnt is drained after compute.
__global__ __launch_bounds__(256) void gemm_bf16_kernel(
    const bf16_t* __restrict__ A, const bf16_t* __restrict__ B,
    const float* __restrict__ bias,
    float* __restrict__ Cf, bf16_t* __restrict__ Cb,
    int M, int N, int K, int relu)
{
  __shared__ bf16_t As[2 * TILE_ELEMS];  // row-major, padded 64B->80B rows
  __shared__ bf16_t Bs[2 * TILE_ELEMS];  // stored transposed: [n][k]
  int tid = threadIdx.x;
  int lane = tid & 31, wid = tid >> 5;
  int waveM = wid >> 1, waveN = wid & 1;
  int rowBase = blockIdx.y * 128;
  int colBase = blockIdx.x * 128;
  int kk = tid >> 3, jb = (tid & 7) * 16;        // B-tile staging coords
#if !USE_TDM
  int ar = tid >> 1, ach = (tid & 1) * 16;       // A-tile staging coords
#endif
  int nt = K >> 5;

  v8f acc[2][4] = {};

  // ---- prologue: stage tile 0 into buffer 0
#if USE_TDM
  if (wid == 0)
    tdm_load_tile_bf16(A + (size_t)rowBase * K, (unsigned int)(uintptr_t)(void*)As,
                       32, 128, K, 3, 3);
#else
  {
    const bf16_t* g = A + (size_t)(rowBase + ar) * K + ach;
    *(v8bf*)(As + ar * 40 + ach) = *(const v8bf*)g;
    *(v8bf*)(As + ar * 40 + ach + 8) = *(const v8bf*)(g + 8);
  }
#endif
  {
    const bf16_t* g = B + (size_t)kk * N + colBase + jb;
    v8bf lo = *(const v8bf*)g;
    v8bf hi = *(const v8bf*)(g + 8);
#pragma unroll
    for (int j = 0; j < 8; ++j) Bs[(jb + j) * 40 + kk] = lo[j];
#pragma unroll
    for (int j = 0; j < 8; ++j) Bs[(jb + 8 + j) * 40 + kk] = hi[j];
  }
#if USE_TDM
  __builtin_amdgcn_s_wait_tensorcnt(0);
#endif
  __syncthreads();

  for (int i = 0; i < nt; ++i) {
    int buf = i & 1, nb = buf ^ 1;
    bool hasNext = (i + 1) < nt;
    const bf16_t* Ac = As + buf * TILE_ELEMS;
    const bf16_t* Bc = Bs + buf * TILE_ELEMS;

    // ---- issue stage i+1 data movement (overlaps with compute below)
    v8bf blo, bhi;
#if !USE_TDM
    v8bf alo, ahi;
#endif
    if (hasNext) {
      int k0 = (i + 1) << 5;
#if USE_TDM
      if (wid == 0)
        tdm_load_tile_bf16(A + (size_t)rowBase * K + k0,
                           (unsigned int)(uintptr_t)(void*)(As + nb * TILE_ELEMS),
                           32, 128, K, 3, 3);
#else
      {
        const bf16_t* g = A + (size_t)(rowBase + ar) * K + k0 + ach;
        alo = *(const v8bf*)g;
        ahi = *(const v8bf*)(g + 8);
      }
#endif
      const bf16_t* g = B + (size_t)(k0 + kk) * N + colBase + jb;
      blo = *(const v8bf*)g;
      bhi = *(const v8bf*)(g + 8);
      if (i + 2 < nt) __builtin_prefetch(g + (size_t)32 * N, 0, 1);
    }

    // ---- compute stage i
    v16bf afr[2], bfr[4];
#pragma unroll
    for (int mi = 0; mi < 2; ++mi)
      afr[mi] = load_frag_k_contig(Ac + (waveM * 32 + mi * 16) * 40, 40, lane);
#pragma unroll
    for (int ni = 0; ni < 4; ++ni)
      bfr[ni] = load_frag_k_contig(Bc + (waveN * 64 + ni * 16) * 40, 40, lane);
#pragma unroll
    for (int mi = 0; mi < 2; ++mi)
#pragma unroll
      for (int ni = 0; ni < 4; ++ni)
        acc[mi][ni] = wmma_bf16(afr[mi], bfr[ni], acc[mi][ni]);

    // ---- drain stage i+1 staging into the alternate buffers
    if (hasNext) {
      bf16_t* Bn = Bs + nb * TILE_ELEMS;
#pragma unroll
      for (int j = 0; j < 8; ++j) Bn[(jb + j) * 40 + kk] = blo[j];
#pragma unroll
      for (int j = 0; j < 8; ++j) Bn[(jb + 8 + j) * 40 + kk] = bhi[j];
#if USE_TDM
      __builtin_amdgcn_s_wait_tensorcnt(0);
#else
      bf16_t* An = As + nb * TILE_ELEMS;
      *(v8bf*)(An + ar * 40 + ach) = alo;
      *(v8bf*)(An + ar * 40 + ach + 8) = ahi;
#endif
    }
    __syncthreads();
  }

  // epilogue (D layout: n = lane&15; m = (lane>>4)*8 + r); uniform-flag
  // branches hoisted, linear pointer walk down rows.
  int nl = lane & 15;
  int mh = (lane >> 4) * 8;
  int row0 = rowBase + waveM * 32 + mh;
  int col0 = colBase + waveN * 64 + nl;
#pragma unroll
  for (int mi = 0; mi < 2; ++mi) {
#pragma unroll
    for (int ni = 0; ni < 4; ++ni) {
      int col = col0 + ni * 16;
      float bv = bias ? bias[col] : 0.0f;
      size_t base = (size_t)(row0 + mi * 16) * N + col;
      if (Cb) {
        bf16_t* p = Cb + base;
#pragma unroll
        for (int r = 0; r < 8; ++r) {
          float v = acc[mi][ni][r] + bv;
          if (relu) v = fmaxf(v, 0.0f);
          *p = (bf16_t)v;
          p += N;
        }
      } else {
        float* p = Cf + base;
#pragma unroll
        for (int r = 0; r < 8; ++r) {
          float v = acc[mi][ni][r] + bv;
          if (relu) v = fmaxf(v, 0.0f);
          *p = v;
          p += N;
        }
      }
    }
  }
}

// ---------------- banded attention (half-width 99) ----------------
// one wave per (16-row q-tile, b, h). QKV: (8192, 2304) bf16, cols [q|k|v].
// E: (NH, 199, 96) bf16. O out: (8192, 768) bf16 at col h*96+a.
__global__ __launch_bounds__(32) void attn_kernel(
    const bf16_t* __restrict__ QKV, const bf16_t* __restrict__ E,
    bf16_t* __restrict__ O)
{
  __shared__ float  Ll[16 * NCOLS];
  __shared__ float  Pl[16 * NCOLS];
  __shared__ bf16_t Pr[16 * NCOLS];
  int lane = threadIdx.x;
  int q0 = blockIdx.x * 16;
  int bh = blockIdx.y;
  int b = bh >> 3, h = bh & 7;
  const int kstart = q0 - 99;
  const int lm = lane & 15;
  const int kh = (lane >> 4) << 3;
  const int mh = (lane >> 4) * 8;

  // Q fragments (3 chunks of K=32 over D_QKV=96)
  v16bf aq[3];
  {
    int t = q0 + lm;
    const bf16_t* qp = QKV + (size_t)(t * BQ + b) * (3 * DM) + h * DH + kh;
#pragma unroll
    for (int c = 0; c < 3; ++c)
      aq[c] = cat8(*(const v8bf*)(qp + c * 32), *(const v8bf*)(qp + c * 32 + 16));
  }

  // content logits: Q * K^T over the band (14 n-tiles)
  for (int j = 0; j < NT; ++j) {
    v8f acc = {};
    int krow = kstart + j * 16 + lm;
    krow = min(max(krow, 0), LQ - 1);
    const bf16_t* kp = QKV + (size_t)(krow * BQ + b) * (3 * DM) + DM + h * DH + kh;
#pragma unroll
    for (int c = 0; c < 3; ++c)
      acc = wmma_bf16(aq[c], cat8(*(const v8bf*)(kp + c * 32), *(const v8bf*)(kp + c * 32 + 16)), acc);
#pragma unroll
    for (int r = 0; r < 8; ++r)
      Ll[(mh + r) * NCOLS + j * 16 + lm] = acc[r];
  }

  // rel-pos logits: Q * E^T (13 n-tiles cover 199 offsets)
  for (int j = 0; j < 13; ++j) {
    v8f acc = {};
    int rr = min(j * 16 + lm, NREL - 1);
    const bf16_t* ep = E + ((size_t)h * NREL + rr) * DH + kh;
#pragma unroll
    for (int c = 0; c < 3; ++c)
      acc = wmma_bf16(aq[c], cat8(*(const v8bf*)(ep + c * 32), *(const v8bf*)(ep + c * 32 + 16)), acc);
#pragma unroll
    for (int r = 0; r < 8; ++r)
      Pl[(mh + r) * NCOLS + j * 16 + lm] = acc[r];
  }
  __syncthreads();

  // band-masked softmax per row (skew gather of rel-pos)
  const float scale = 0.10206207261596577f; // 1/sqrt(96)
  for (int m = 0; m < 16; ++m) {
    int t = q0 + m;
    float lv[7];
    float mx = -3.0e38f;
#pragma unroll
    for (int i = 0; i < 7; ++i) {
      int c = lane + i * 32;
      int k = kstart + c;
      int r = k - t + 99;
      bool valid = (k >= 0) && (k < LQ) && (r >= 0) && (r < NREL);
      float v = valid ? (Ll[m * NCOLS + c] * scale + Pl[m * NCOLS + r]) : -1.0e30f;
      lv[i] = v;
      mx = fmaxf(mx, v);
    }
#pragma unroll
    for (int s = 16; s > 0; s >>= 1) mx = fmaxf(mx, __shfl_xor(mx, s, 32));
    float sum = 0.f;
#pragma unroll
    for (int i = 0; i < 7; ++i) { lv[i] = __expf(lv[i] - mx); sum += lv[i]; }
#pragma unroll
    for (int s = 16; s > 0; s >>= 1) sum += __shfl_xor(sum, s, 32);
    float inv = 1.0f / sum;
#pragma unroll
    for (int i = 0; i < 7; ++i)
      Pr[m * NCOLS + lane + i * 32] = (bf16_t)(lv[i] * inv);
  }
  __syncthreads();

  // O = probs * V  (6 n-tiles of 16 over D_QKV, 7 K-chunks of 32 over band)
  for (int ni = 0; ni < 6; ++ni) {
    v8f acc = {};
    for (int kc = 0; kc < 7; ++kc) {
      const bf16_t* pp = Pr + lm * NCOLS + kc * 32 + kh;
      v16bf afr = cat8(*(const v8bf*)pp, *(const v8bf*)(pp + 16));
      v16bf bfr;
      int coln = 2 * DM + h * DH + ni * 16 + lm;
#pragma unroll
      for (int j = 0; j < 16; ++j) {
        int klocal = (j < 8) ? (kh + j) : (16 + kh + (j - 8));
        int k = kstart + kc * 32 + klocal;
        k = min(max(k, 0), LQ - 1);
        bfr[j] = QKV[(size_t)(k * BQ + b) * (3 * DM) + coln];
      }
      acc = wmma_bf16(afr, bfr, acc);
    }
#pragma unroll
    for (int r = 0; r < 8; ++r) {
      int t = q0 + mh + r;
      O[(size_t)(t * BQ + b) * DM + h * DH + ni * 16 + lm] = (bf16_t)acc[r];
    }
  }
}

// ---------------- fused residual + LayerNorm ----------------
__global__ __launch_bounds__(256) void add_ln_kernel(
    const float* __restrict__ X, const float* __restrict__ Y,
    const float* __restrict__ g, const float* __restrict__ be,
    float* __restrict__ outF, bf16_t* __restrict__ outB)
{
  __shared__ float red[256];
  int row = blockIdx.x;
  const float* xr = X + (size_t)row * DM;
  const float* yr = Y + (size_t)row * DM;
  float s[3];
  float lsum = 0.f;
#pragma unroll
  for (int i = 0; i < 3; ++i) {
    int c = threadIdx.x + i * 256;
    s[i] = xr[c] + yr[c];
    lsum += s[i];
  }
  red[threadIdx.x] = lsum; __syncthreads();
  for (int st = 128; st > 0; st >>= 1) {
    if (threadIdx.x < st) red[threadIdx.x] += red[threadIdx.x + st];
    __syncthreads();
  }
  float mu = red[0] * (1.0f / DM);
  __syncthreads();
  float lvar = 0.f;
#pragma unroll
  for (int i = 0; i < 3; ++i) { float d = s[i] - mu; lvar += d * d; }
  red[threadIdx.x] = lvar; __syncthreads();
  for (int st = 128; st > 0; st >>= 1) {
    if (threadIdx.x < st) red[threadIdx.x] += red[threadIdx.x + st];
    __syncthreads();
  }
  float rstd = rsqrtf(red[0] * (1.0f / DM) + 1e-5f);
#pragma unroll
  for (int i = 0; i < 3; ++i) {
    int c = threadIdx.x + i * 256;
    float v = (s[i] - mu) * rstd * g[c] + be[c];
    if (outF) outF[(size_t)row * DM + c] = v;
    if (outB) outB[(size_t)row * DM + c] = (bf16_t)v;
  }
}

// ---------------- launch ----------------
extern "C" void kernel_launch(void* const* d_in, const int* in_sizes, int n_in,
                              void* d_out, int out_size, void* d_ws, size_t ws_size,
                              hipStream_t stream) {
  (void)in_sizes; (void)n_in; (void)out_size; (void)ws_size;
  const float* x     = (const float*)d_in[0];
  const float* w_q   = (const float*)d_in[1];
  const float* w_k   = (const float*)d_in[2];
  const float* w_v   = (const float*)d_in[3];
  const float* w_o   = (const float*)d_in[4];
  const float* rel   = (const float*)d_in[5];
  const float* w1    = (const float*)d_in[6];
  const float* b1    = (const float*)d_in[7];
  const float* w2    = (const float*)d_in[8];
  const float* b2    = (const float*)d_in[9];
  const float* ln1g  = (const float*)d_in[10];
  const float* ln1b  = (const float*)d_in[11];
  const float* ln2g  = (const float*)d_in[12];
  const float* ln2b  = (const float*)d_in[13];
  float* out = (float*)d_out;

  const int M = LQ * BQ;            // 8192 token rows
  char* ws = (char*)d_ws;
  size_t off = 0;
  auto take = [&](size_t bytes) -> char* {
    char* p = ws + off;
    off = (off + bytes + 255) & ~(size_t)255;
    return p;
  };
  bf16_t* Xbf   = (bf16_t*)take((size_t)M * DM * 2);
  bf16_t* Wqkv  = (bf16_t*)take((size_t)DM * 3 * DM * 2);
  bf16_t* Wo    = (bf16_t*)take((size_t)DM * DM * 2);
  bf16_t* W1b   = (bf16_t*)take((size_t)DM * DFF * 2);
  bf16_t* W2b   = (bf16_t*)take((size_t)DFF * DM * 2);
  bf16_t* Eb    = (bf16_t*)take((size_t)NH * NREL * DH * 2);
  bf16_t* QKV   = (bf16_t*)take((size_t)M * 3 * DM * 2);
  bf16_t* Obuf  = (bf16_t*)take((size_t)M * DM * 2);
  float*  attnF = (float*)take((size_t)M * DM * 4);
  float*  srcF  = (float*)take((size_t)M * DM * 4);
  bf16_t* srcB  = (bf16_t*)take((size_t)M * DM * 2);
  bf16_t* Hb    = (bf16_t*)take((size_t)M * DFF * 2);
  float*  ffF   = (float*)take((size_t)M * DM * 4);

  auto cblk = [](int n) { return dim3((n + 255) / 256); };

  // packs
  conv_f32_bf16<<<cblk(M * DM), 256, 0, stream>>>(x, Xbf, M * DM);
  conv_f32_bf16<<<cblk(DM * DM), 256, 0, stream>>>(w_o, Wo, DM * DM);
  conv_f32_bf16<<<cblk(DM * DFF), 256, 0, stream>>>(w1, W1b, DM * DFF);
  conv_f32_bf16<<<cblk(DFF * DM), 256, 0, stream>>>(w2, W2b, DFF * DM);
  conv_f32_bf16<<<cblk(NH * NREL * DH), 256, 0, stream>>>(rel, Eb, NH * NREL * DH);
  pack_qkv_w<<<cblk(DM * 3 * DM), 256, 0, stream>>>(w_q, w_k, w_v, Wqkv);

  // QKV projection: (8192 x 768) x (768 x 2304) -> bf16
  gemm_bf16_kernel<<<dim3(3 * DM / 128, M / 128), 256, 0, stream>>>(
      Xbf, Wqkv, nullptr, nullptr, QKV, M, 3 * DM, DM, 0);

  // banded attention with rel-pos
  attn_kernel<<<dim3(LQ / 16, BQ * NH), 32, 0, stream>>>(QKV, Eb, Obuf);

  // output projection: (8192 x 768) x (768 x 768) -> f32
  gemm_bf16_kernel<<<dim3(DM / 128, M / 128), 256, 0, stream>>>(
      Obuf, Wo, nullptr, attnF, nullptr, M, DM, DM, 0);

  // src = LN1(x + attn)
  add_ln_kernel<<<dim3(M), 256, 0, stream>>>(x, attnF, ln1g, ln1b, srcF, srcB);

  // FFN1: relu(src @ w1 + b1) -> bf16
  gemm_bf16_kernel<<<dim3(DFF / 128, M / 128), 256, 0, stream>>>(
      srcB, W1b, b1, nullptr, Hb, M, DFF, DM, 1);

  // FFN2: h @ w2 + b2 -> f32
  gemm_bf16_kernel<<<dim3(DM / 128, M / 128), 256, 0, stream>>>(
      Hb, W2b, b2, ffF, nullptr, M, DM, DFF, 0);

  // out = LN2(src + ff)
  add_ln_kernel<<<dim3(M), 256, 0, stream>>>(srcF, ffF, ln2g, ln2b, out, nullptr);
}
